// GatedAttentionUnit_79809082294687
// MI455X (gfx1250) — matrix operand order repaired and checked by reading, not compiled, not run on hardware
//
#include <hip/hip_runtime.h>
#include <hip/hip_bf16.h>

// GAU: B=8 N=2048 D=256 S=128 H=512. All matmuls via v_wmma_f32_16x16x32_f16.
// Intermediates (u,v,q,k,gated) kept in f16 in d_ws. Attention tiled 128x128
// (relu^2 "softmax" is local -> no running stats), double-buffered K/V tiles
// staged with GLOBAL_LOAD_ASYNC_TO_LDS_B128 (ASYNCcnt path).

typedef __attribute__((ext_vector_type(16))) _Float16 v16h;
typedef __attribute__((ext_vector_type(8)))  _Float16 v8h;
typedef __attribute__((ext_vector_type(8)))  float    v8f;
typedef __attribute__((ext_vector_type(4)))  int      v4i_t;

#define Bn 8
#define Nn 2048
#define Dn 256
#define Sn 128
#define Hn 512
#define En 1152        // 2H+S
#define TOK (Bn*Nn)    // 16384

#ifndef __has_builtin
#define __has_builtin(x) 0
#endif
#if __has_builtin(__builtin_amdgcn_global_load_async_to_lds_b128)
#define HAVE_ASYNC 1
#else
#define HAVE_ASYNC 0
#endif

#if HAVE_ASYNC
typedef __attribute__((address_space(1))) v4i_t as1_v4i;
typedef __attribute__((address_space(3))) v4i_t as3_v4i;
// One 16B lane-copy global->LDS on the ASYNCcnt path (no VGPR round trip).
static __device__ __forceinline__ void cp16_async(_Float16* dst, const _Float16* src) {
  __builtin_amdgcn_global_load_async_to_lds_b128((as1_v4i*)src, (as3_v4i*)dst, 0, 0);
}
#endif
static __device__ __forceinline__ void async_acquire() {
#if HAVE_ASYNC
  asm volatile("s_wait_asynccnt 0x0" ::: "memory");
#endif
}

static __device__ __forceinline__ v8f v8f_zero() {
  v8f z = {0.f,0.f,0.f,0.f,0.f,0.f,0.f,0.f};
  return z;
}

// A fragment, 16x32 f16 (row-major tile, ld in halfs). ISA 7.12.2 layout:
// lane m=l&15, group g=l>>4; halves 0..7 -> K=g*8+i, halves 8..15 -> K=16+g*8+i.
static __device__ __forceinline__ v16h frag_a16(const _Float16* base, int ld) {
  const int lane = threadIdx.x & 31;
  const _Float16* p = base + (lane & 15) * ld + ((lane >> 4) << 3);
  v8h lo = *(const v8h*)(p);
  v8h hi = *(const v8h*)(p + 16);
  v16h r;
#pragma unroll
  for (int i = 0; i < 8; ++i) { r[i] = lo[i]; r[i + 8] = hi[i]; }
  return r;
}

// B fragment, 32x16 f16 from an N-major tile (baseT[n*ld + k]):
// lane n=l&15, group g=l>>4; halves 0..7 -> K=g*8+i, halves 8..15 -> K=16+g*8+i.
static __device__ __forceinline__ v16h frag_b16(const _Float16* baseT, int ld) {
  const int lane = threadIdx.x & 31;
  const _Float16* p = baseT + (lane & 15) * ld + ((lane >> 4) << 3);
  v8h lo = *(const v8h*)(p);
  v8h hi = *(const v8h*)(p + 16);
  v16h r;
#pragma unroll
  for (int i = 0; i < 8; ++i) { r[i] = lo[i]; r[i + 8] = hi[i]; }
  return r;
}

static __device__ __forceinline__ v8f wmma16(v16h a, v16h b, v8f c) {
  return __builtin_amdgcn_wmma_f32_16x16x32_f16(false, a, false, b, (short)0, c,
                                                false, false);
}

// ---------------------------------------------------------------- k_prep ----
// f32 weights -> f16, transposed to N-major (K contiguous) for B fragments.
__global__ void k_prep(const float* __restrict__ W_uv, const float* __restrict__ W_o,
                       _Float16* __restrict__ wuvT, _Float16* __restrict__ woT) {
  int idx = blockIdx.x * 256 + threadIdx.x;
  const int n_uv = Dn * En;   // 294912
  const int n_o  = Hn * Dn;   // 131072
  if (idx < n_uv) {
    int d = idx / En, e = idx - d * En;
    wuvT[e * Dn + d] = (_Float16)W_uv[idx];        // wuvT[e][d]
  } else if (idx < n_uv + n_o) {
    int j = idx - n_uv;
    int h = j / Dn, d = j - h * Dn;
    woT[d * Hn + h] = (_Float16)W_o[j];            // woT[d][h]
  }
}

// ---------------------------------------------------------------- k_proj ----
// 64 rows/block: RMSNorm then xn(64x256) @ W_uv(256x1152) in 9 chunks of 128
// cols; fused bias+silu epilogue scatters u, pre-transposed v, and
// gamma/beta-scaled q,k (f16).
#define PLD 264   // 256 + 8 pad (16B-aligned rows, bank spread)
__global__ void __launch_bounds__(256) k_proj(
    const float* __restrict__ x, const float* __restrict__ gp,
    const _Float16* __restrict__ wuvT, const float* __restrict__ b_uv,
    const float* __restrict__ gamma, const float* __restrict__ beta,
    _Float16* __restrict__ u, _Float16* __restrict__ vTg,
    _Float16* __restrict__ q, _Float16* __restrict__ k) {
  extern __shared__ __align__(16) unsigned char smem_raw[];
  float*    red = (float*)smem_raw;                 // 256 f32
  _Float16* xn  = (_Float16*)(smem_raw + 1024);     // 64 x PLD
  _Float16* wt  = xn + 64 * PLD;                    // 128 x PLD

  const int t = threadIdx.x;
  const int token0 = blockIdx.x * 64;
  const float gscale = gp[0];

  { // RMSNorm: 4 threads per row, 64 f32 each
    const int row = t >> 2, seg = t & 3;
    const float* xr = x + (size_t)(token0 + row) * Dn + seg * 64;
    float4 f4[16];
    float ss = 0.f;
#pragma unroll
    for (int i = 0; i < 16; ++i) {
      f4[i] = ((const float4*)xr)[i];
      ss += f4[i].x*f4[i].x + f4[i].y*f4[i].y + f4[i].z*f4[i].z + f4[i].w*f4[i].w;
    }
    red[t] = ss;
    __syncthreads();
    float tot = red[row*4+0] + red[row*4+1] + red[row*4+2] + red[row*4+3];
    float rms = __builtin_sqrtf(tot * (1.0f / Dn));
    float sc  = gscale / fmaxf(rms, 1e-5f);
    _Float16* xrow = xn + row * PLD + seg * 64;
#pragma unroll
    for (int i = 0; i < 16; ++i) {
      xrow[i*4+0] = (_Float16)(f4[i].x * sc);
      xrow[i*4+1] = (_Float16)(f4[i].y * sc);
      xrow[i*4+2] = (_Float16)(f4[i].z * sc);
      xrow[i*4+3] = (_Float16)(f4[i].w * sc);
    }
  }

  const int lane = t & 31, w = t >> 5;
  const int strip = w & 3, half = w >> 2;   // 4 M-strips x 2 N-halves

  for (int c = 0; c < 9; ++c) {
    __syncthreads();
    { // stage 128 W^T rows (K contiguous) into LDS
      const int el = t >> 1, seg2 = t & 1;
      const _Float16* src = wuvT + (size_t)(c * 128 + el) * Dn + seg2 * 128;
      _Float16* dst = wt + el * PLD + seg2 * 128;
#if HAVE_ASYNC
#pragma unroll
      for (int i = 0; i < 16; ++i) cp16_async(dst + i * 8, src + i * 8);
#else
      v8h tmp[16];
#pragma unroll
      for (int i = 0; i < 16; ++i) tmp[i] = *(const v8h*)(src + i * 8);
#pragma unroll
      for (int i = 0; i < 16; ++i) *(v8h*)(dst + i * 8) = tmp[i];
#endif
      if (c + 1 < 9)
        __builtin_prefetch(wuvT + (size_t)((c + 1) * 128 + el) * Dn + seg2 * 128, 0, 1);
    }
    async_acquire();
    __syncthreads();

    v8f acc[4];
#pragma unroll
    for (int i = 0; i < 4; ++i) acc[i] = v8f_zero();
#pragma unroll
    for (int kk = 0; kk < 8; ++kk) {
      v16h a = frag_a16(xn + strip * 16 * PLD + kk * 32, PLD);
#pragma unroll
      for (int nt = 0; nt < 4; ++nt) {
        v16h b = frag_b16(wt + (half * 64 + nt * 16) * PLD + kk * 32, PLD);
        acc[nt] = wmma16(a, b, acc[nt]);
      }
    }
    // epilogue: bias + silu, scatter u/v/q/k (branch uniform per 16-col tile)
#pragma unroll
    for (int nt = 0; nt < 4; ++nt) {
      const int e = c * 128 + half * 64 + nt * 16 + (lane & 15);
#pragma unroll
      for (int r = 0; r < 8; ++r) {
        const int token = token0 + strip * 16 + r + ((lane >> 4) << 3);
        float val = acc[nt][r] + b_uv[e];
        val = val / (1.0f + __expf(-val));          // silu
        if (e < Hn) {
          u[(size_t)token * Hn + e] = (_Float16)val;
        } else if (e < 2 * Hn) {
          // v stored pre-transposed: vTg[b][h][n] so k_attn B-stages are row copies
          const int b = token >> 11, n = token & (Nn - 1);
          vTg[((size_t)b * Hn + (e - Hn)) * Nn + n] = (_Float16)val;
        } else {
          const int s = e - 2 * Hn;
          q[(size_t)token * Sn + s] = (_Float16)(val * gamma[s]      + beta[s]);
          k[(size_t)token * Sn + s] = (_Float16)(val * gamma[Sn + s] + beta[Sn + s]);
        }
      }
    }
  }
}

// ---------------------------------------------------------------- k_attn ----
// grid (B, N/128, H/128). Block owns 128 q-rows x 128 H-cols; streams 16
// K/V tiles of 128 tokens with double-buffered async staging. Per wave:
// qk strip (32 WMMA) -> relu^2 -> A-repack via private LDS strip ->
// attn@v (32 WMMA). Epilogue fuses the u gate.
#define ALD 136   // 128 + 8 pad
__global__ void __launch_bounds__(256) k_attn(
    const _Float16* __restrict__ q, const _Float16* __restrict__ kmat,
    const _Float16* __restrict__ vTg, const _Float16* __restrict__ u,
    _Float16* __restrict__ gated) {
  extern __shared__ __align__(16) unsigned char smem_raw[];
  _Float16* qlds = (_Float16*)smem_raw;      // 128 x ALD
  _Float16* kbuf[2] = {qlds + 128 * ALD, qlds + 2 * 128 * ALD};
  _Float16* vbuf[2] = {qlds + 3 * 128 * ALD, qlds + 4 * 128 * ALD};
  _Float16* alds = qlds + 5 * 128 * ALD;     // 8 private wave strips

  const int t = threadIdx.x, lane = t & 31, w = t >> 5;
  const int bI = blockIdx.x;
  const int tokenI0 = bI * Nn + blockIdx.y * 128;
  const int h0 = blockIdx.z * 128;
  const float rs = 0.08838834764831845f;     // rsqrt(S=128)

  const int row = t >> 1, seg = t & 1;       // staging split: 2 thr/row, 64 halfs

  // stage K tile jj (token-major) + V tile jj (h-major, pre-transposed in HBM)
  auto stageKV = [&](int jj, int bi) {
    {
      const _Float16* src = kmat + (size_t)(bI * Nn + jj * 128 + row) * Sn + seg * 64;
      _Float16* dst = kbuf[bi] + row * ALD + seg * 64;
#if HAVE_ASYNC
#pragma unroll
      for (int i = 0; i < 8; ++i) cp16_async(dst + i * 8, src + i * 8);
#else
      v8h tmp[8];
#pragma unroll
      for (int i = 0; i < 8; ++i) tmp[i] = *(const v8h*)(src + i * 8);
#pragma unroll
      for (int i = 0; i < 8; ++i) *(v8h*)(dst + i * 8) = tmp[i];
#endif
    }
    {
      const _Float16* src = vTg + ((size_t)bI * Hn + h0 + row) * Nn + jj * 128 + seg * 64;
      _Float16* dst = vbuf[bi] + row * ALD + seg * 64;
#if HAVE_ASYNC
#pragma unroll
      for (int i = 0; i < 8; ++i) cp16_async(dst + i * 8, src + i * 8);
#else
      v8h tmp[8];
#pragma unroll
      for (int i = 0; i < 8; ++i) tmp[i] = *(const v8h*)(src + i * 8);
#pragma unroll
      for (int i = 0; i < 8; ++i) *(v8h*)(dst + i * 8) = tmp[i];
#endif
    }
  };

  { // stage q tile
    const _Float16* src = q + (size_t)(tokenI0 + row) * Sn + seg * 64;
    _Float16* dst = qlds + row * ALD + seg * 64;
#if HAVE_ASYNC
#pragma unroll
    for (int i = 0; i < 8; ++i) cp16_async(dst + i * 8, src + i * 8);
#else
    v8h tmp[8];
#pragma unroll
    for (int i = 0; i < 8; ++i) tmp[i] = *(const v8h*)(src + i * 8);
#pragma unroll
    for (int i = 0; i < 8; ++i) *(v8h*)(dst + i * 8) = tmp[i];
#endif
  }
  stageKV(0, 0);
  async_acquire();
  __syncthreads();

  v16h aq[4];
#pragma unroll
  for (int c = 0; c < 4; ++c) aq[c] = frag_a16(qlds + w * 16 * ALD + c * 32, ALD);

  v8f gacc[8];
#pragma unroll
  for (int i = 0; i < 8; ++i) gacc[i] = v8f_zero();

  for (int j = 0; j < 16; ++j) {
    const int cur = j & 1;
    if (j > 0) {            // tile j staged last iter; all waves done with j-1
      async_acquire();
      __syncthreads();
    }
    if (j + 1 < 16) stageKV(j + 1, cur ^ 1);   // prefetch into free buffer

    // qk strip 16x128, relu^2, repack into private A strip
#pragma unroll
    for (int nt2 = 0; nt2 < 8; ++nt2) {
      v8f qk = v8f_zero();
#pragma unroll
      for (int c = 0; c < 4; ++c) {
        v16h bf = frag_b16(kbuf[cur] + (nt2 * 16) * ALD + c * 32, ALD);
        qk = wmma16(aq[c], bf, qk);
      }
#pragma unroll
      for (int r = 0; r < 8; ++r) {
        float a = qk[r] * rs;
        a = a > 0.f ? a * a : 0.f;
        const int m = r + ((lane >> 4) << 3);
        alds[(w * 16 + m) * ALD + nt2 * 16 + (lane & 15)] = (_Float16)a;
      }
    }
    // gated += attn(16x128) @ V(128x128)
#pragma unroll
    for (int c = 0; c < 4; ++c) {
      v16h a2 = frag_a16(alds + w * 16 * ALD + c * 32, ALD);
#pragma unroll
      for (int nt = 0; nt < 8; ++nt) {
        v16h bf = frag_b16(vbuf[cur] + (nt * 16) * ALD + c * 32, ALD);
        gacc[nt] = wmma16(a2, bf, gacc[nt]);
      }
    }
  }

  // epilogue: gate by u, store f16 gated
#pragma unroll
  for (int nt = 0; nt < 8; ++nt) {
    const int hc = h0 + nt * 16 + (lane & 15);
#pragma unroll
    for (int r = 0; r < 8; ++r) {
      const int token = tokenI0 + w * 16 + r + ((lane >> 4) << 3);
      float uval = (float)u[(size_t)token * Hn + hc];
      gated[(size_t)token * Hn + hc] = (_Float16)(gacc[nt][r] * uval);
    }
  }
}

// ----------------------------------------------------------------- k_out ----
// out(64x256) = gated(64x512) @ W_o(512x256) + b_o, K streamed in 128-chunks.
#define OLD 136
__global__ void __launch_bounds__(256) k_out(
    const _Float16* __restrict__ gated, const _Float16* __restrict__ woT,
    const float* __restrict__ b_o, float* __restrict__ out) {
  extern __shared__ __align__(16) unsigned char smem_raw[];
  _Float16* glds = (_Float16*)smem_raw;     // 64 x OLD
  _Float16* wlds = glds + 64 * OLD;         // 256 x OLD

  const int t = threadIdx.x, lane = t & 31, w = t >> 5;
  const int strip = w & 3, half = w >> 2;
  const int token0 = blockIdx.x * 64;

  v8f acc[8];
#pragma unroll
  for (int i = 0; i < 8; ++i) acc[i] = v8f_zero();

  for (int kc = 0; kc < 4; ++kc) {
    __syncthreads();
    { // stage gated K-chunk
      const int row = t >> 2, seg = t & 3;
      const _Float16* src = gated + (size_t)(token0 + row) * Hn + kc * 128 + seg * 32;
      _Float16* dst = glds + row * OLD + seg * 32;
#if HAVE_ASYNC
#pragma unroll
      for (int i = 0; i < 4; ++i) cp16_async(dst + i * 8, src + i * 8);
#else
      v8h tmp[4];
#pragma unroll
      for (int i = 0; i < 4; ++i) tmp[i] = *(const v8h*)(src + i * 8);
#pragma unroll
      for (int i = 0; i < 4; ++i) *(v8h*)(dst + i * 8) = tmp[i];
#endif
    }
    { // stage W_o^T K-chunk: one d-row per thread
      const _Float16* src = woT + (size_t)t * Hn + kc * 128;
      _Float16* dst = wlds + t * OLD;
#if HAVE_ASYNC
#pragma unroll
      for (int i = 0; i < 16; ++i) cp16_async(dst + i * 8, src + i * 8);
#else
      v8h tmp[16];
#pragma unroll
      for (int i = 0; i < 16; ++i) tmp[i] = *(const v8h*)(src + i * 8);
#pragma unroll
      for (int i = 0; i < 16; ++i) *(v8h*)(dst + i * 8) = tmp[i];
#endif
      if (kc + 1 < 4)
        __builtin_prefetch(woT + (size_t)t * Hn + (kc + 1) * 128, 0, 1);
    }
    async_acquire();
    __syncthreads();
#pragma unroll
    for (int c = 0; c < 4; ++c) {
      v16h a = frag_a16(glds + strip * 16 * OLD + c * 32, OLD);
#pragma unroll
      for (int nt = 0; nt < 8; ++nt) {
        v16h bf = frag_b16(wlds + (half * 128 + nt * 16) * OLD + c * 32, OLD);
        acc[nt] = wmma16(a, bf, acc[nt]);
      }
    }
  }
#pragma unroll
  for (int nt = 0; nt < 8; ++nt) {
    const int d = half * 128 + nt * 16 + (lane & 15);
    const float bo = b_o[d];
#pragma unroll
    for (int r = 0; r < 8; ++r) {
      const int token = token0 + strip * 16 + r + ((lane >> 4) << 3);
      out[(size_t)token * Dn + d] = acc[nt][r] + bo;
    }
  }
}

// ----------------------------------------------------------------- launch ---
extern "C" void kernel_launch(void* const* d_in, const int* in_sizes, int n_in,
                              void* d_out, int out_size, void* d_ws, size_t ws_size,
                              hipStream_t stream) {
  const float* x     = (const float*)d_in[0];
  const float* g     = (const float*)d_in[1];
  const float* W_uv  = (const float*)d_in[2];
  const float* b_uv  = (const float*)d_in[3];
  const float* gamma = (const float*)d_in[4];
  const float* beta  = (const float*)d_in[5];
  const float* W_o   = (const float*)d_in[6];
  const float* b_o   = (const float*)d_in[7];
  float* out = (float*)d_out;

  _Float16* ws    = (_Float16*)d_ws;
  _Float16* wuvT  = ws;                                  // 1152*256
  _Float16* woT   = wuvT + (size_t)En * Dn;              // 256*512
  _Float16* u     = woT  + (size_t)Dn * Hn;              // TOK*H
  _Float16* vTg   = u    + (size_t)TOK * Hn;             // B*H*N (transposed v)
  _Float16* q     = vTg  + (size_t)TOK * Hn;             // TOK*S
  _Float16* k     = q    + (size_t)TOK * Sn;
  _Float16* gated = k    + (size_t)TOK * Sn;             // TOK*H

  const size_t projLds = 1024 + (size_t)(64 + 128) * PLD * 2;   // ~102 KB
  const size_t attnLds = (size_t)6 * 128 * ALD * 2;             // ~204 KB
  const size_t outLds  = (size_t)(64 + 256) * OLD * 2;          // ~87 KB

  (void)hipFuncSetAttribute((const void*)k_proj, hipFuncAttributeMaxDynamicSharedMemorySize, (int)projLds);
  (void)hipFuncSetAttribute((const void*)k_attn, hipFuncAttributeMaxDynamicSharedMemorySize, (int)attnLds);
  (void)hipFuncSetAttribute((const void*)k_out,  hipFuncAttributeMaxDynamicSharedMemorySize, (int)outLds);

  k_prep<<<dim3((Dn * En + Hn * Dn + 255) / 256), dim3(256), 0, stream>>>(W_uv, W_o, wuvT, woT);
  k_proj<<<dim3(TOK / 64), dim3(256), projLds, stream>>>(x, g, wuvT, b_uv, gamma, beta, u, vTg, q, k);
  k_attn<<<dim3(Bn, Nn / 128, Hn / 128), dim3(256), attnLds, stream>>>(q, k, vTg, u, gated);
  k_out<<<dim3(TOK / 64), dim3(256), outLds, stream>>>(gated, woT, b_o, out);
}
